// ExternalAttention_v3_68607807586379
// MI455X (gfx1250) — matrix operand order repaired and checked by reading, not compiled, not run on hardware
//
#include <hip/hip_runtime.h>
#include <math.h>

#define B_ 16
#define C_ 256
#define N_ 4096
#define K_ 128

typedef __attribute__((ext_vector_type(16))) __bf16 v16bf;
typedef __attribute__((ext_vector_type(8)))  float  v8f;

union Frag { v16bf v; uint4 q[2]; };

// ---------------------------------------------------------------------------
// Tiled transpose + f32->bf16 convert:  src [R, Cc] f32  ->  dst [Cc, R] bf16
// ---------------------------------------------------------------------------
__global__ __launch_bounds__(256) void tconv_bf16(const float* __restrict__ src,
                                                  __bf16* __restrict__ dst,
                                                  int R, int Cc,
                                                  long sBatch, long dBatch) {
  __shared__ float tile[32][33];
  const float* s = src + (size_t)blockIdx.z * sBatch;
  __bf16* d = dst + (size_t)blockIdx.z * dBatch;
  const int c0 = blockIdx.x * 32;   // column in src
  const int r0 = blockIdx.y * 32;   // row in src
  const int tx = threadIdx.x, ty = threadIdx.y;
#pragma unroll
  for (int j = 0; j < 4; ++j)
    tile[ty + 8 * j][tx] = s[(size_t)(r0 + ty + 8 * j) * Cc + c0 + tx];
  __syncthreads();
#pragma unroll
  for (int j = 0; j < 4; ++j)
    d[(size_t)(c0 + ty + 8 * j) * R + r0 + tx] = (__bf16)tile[tx][ty + 8 * j];
}

// ---------------------------------------------------------------------------
// Plain f32 -> bf16 convert (row-major copy)
// ---------------------------------------------------------------------------
__global__ void f32_to_bf16(const float* __restrict__ src, __bf16* __restrict__ dst, int n) {
  int i = blockIdx.x * blockDim.x + threadIdx.x;
  if (i < n) dst[i] = (__bf16)src[i];
}

__global__ void zero_f32(float* __restrict__ p, int n) {
  int i = blockIdx.x * blockDim.x + threadIdx.x;
  if (i < n) p[i] = 0.f;
}

// ---------------------------------------------------------------------------
// Async global->LDS helpers (CDNA5 ASYNCcnt path).
// Moves 32 bytes per lane: two b128 transfers sharing base regs, INST_OFFSET
// applies to both the global source and the LDS destination.
// ---------------------------------------------------------------------------
__device__ __forceinline__ void async_copy32(unsigned ldsOff, const void* gptr) {
  asm volatile(
      "global_load_async_to_lds_b128 %0, %1, off\n\t"
      "global_load_async_to_lds_b128 %0, %1, off offset:16"
      :
      : "v"(ldsOff), "v"(gptr)
      : "memory");
}

// ---------------------------------------------------------------------------
// Generic bf16 WMMA GEMM:  D[b, m, n] = sum_k A[b, m, k] * Bt[n, k]  (+bias)
//   A : [B, M=N_, KD] bf16 row-major       Bt : [ND, KD] bf16 row-major
// Block tile 128(m) x 128(n), K-step 32, double-buffered LDS fed by
// global_load_async_to_lds_b128. 8 waves: wm in 0..3 (32 rows), wn in 0..1
// (64 cols); each wave owns 2x4 v_wmma_f32_16x16x32_bf16 accumulators.
// EPI: 0 = bf16 out            1 = bf16 out + bias
//      2 = f32 out             3 = f32 out + bias + per-channel sum/sumsq atomics
// ---------------------------------------------------------------------------
template <int KD, int EPI>
__global__ __launch_bounds__(256) void gemm_bf16_wmma(
    const __bf16* __restrict__ A, const __bf16* __restrict__ Bt,
    const float* __restrict__ bias, void* __restrict__ D,
    float* __restrict__ csum, float* __restrict__ csumsq, int ND) {
  __shared__ __bf16 lA[2][128 * 32];
  __shared__ __bf16 lB[2][128 * 32];

  const int t    = threadIdx.x;
  const int lane = t & 31;
  const int wave = t >> 5;
  const int wm   = wave & 3;
  const int wn   = wave >> 2;
  const int m0   = blockIdx.x * 128;
  const int n0   = blockIdx.y * 128;
  const size_t batchA = (size_t)blockIdx.z * (size_t)N_ * KD;
  const size_t batchD = (size_t)blockIdx.z * (size_t)N_ * ND;

  // cooperative tile-load coordinates: both tiles are 128 rows x 32 halfs,
  // each of 256 threads moves one 32-byte row chunk per tile.
  const int trow = t >> 1, tcol = (t & 1) * 16;
  const __bf16* gA = A + batchA + (size_t)(m0 + trow) * KD + tcol;
  const __bf16* gB = Bt + (size_t)(n0 + trow) * KD + tcol;

  v8f acc[2][4];
#pragma unroll
  for (int i = 0; i < 2; ++i)
#pragma unroll
    for (int j = 0; j < 4; ++j)
      acc[i][j] = (v8f){0.f, 0.f, 0.f, 0.f, 0.f, 0.f, 0.f, 0.f};

  constexpr int STEPS = KD / 32;

  // prologue: stage k0 = 0 into buffer 0
  async_copy32((unsigned)(size_t)&lA[0][trow * 32 + tcol], gA);
  async_copy32((unsigned)(size_t)&lB[0][trow * 32 + tcol], gB);

#pragma unroll
  for (int s = 0; s < STEPS; ++s) {
    const int k0  = s * 32;
    const int buf = s & 1;

    // all waves done reading buf^1 from step s-1 before we overwrite it
    __syncthreads();
    if (s + 1 < STEPS) {
      async_copy32((unsigned)(size_t)&lA[buf ^ 1][trow * 32 + tcol], gA + k0 + 32);
      async_copy32((unsigned)(size_t)&lB[buf ^ 1][trow * 32 + tcol], gB + k0 + 32);
      if (s + 2 < STEPS) {   // deep prefetch of the tile after next into L2
        __builtin_prefetch(gA + k0 + 64, 0, 1);
        __builtin_prefetch(gB + k0 + 64, 0, 1);
      }
      asm volatile("s_wait_asynccnt 0x4" ::: "memory");  // my loads for buf done
    } else {
      asm volatile("s_wait_asynccnt 0x0" ::: "memory");
    }
    __syncthreads();  // everyone's buf data visible

    // fragment assembly per ISA wave32 layouts:
    //  A 16x32: lanes 0-15 row M=lane hold K {0..7,16..23}; lanes 16-31 K {8..15,24..31}
    //  B 32x16: lanes 0-15 col N=lane hold K {0..7,16..23}; lanes 16-31 K {8..15,24..31}
    const int r  = lane & 15;
    const int hs = lane >> 4;
    Frag af[2], bfr[4];
#pragma unroll
    for (int i = 0; i < 2; ++i) {
      const int row = wm * 32 + i * 16 + r;
      af[i].q[0] = *(const uint4*)&lA[buf][row * 32 + hs * 8];
      af[i].q[1] = *(const uint4*)&lA[buf][row * 32 + 16 + hs * 8];
    }
#pragma unroll
    for (int j = 0; j < 4; ++j) {
      const int nr = wn * 64 + j * 16 + r;
      bfr[j].q[0] = *(const uint4*)&lB[buf][nr * 32 + hs * 8];
      bfr[j].q[1] = *(const uint4*)&lB[buf][nr * 32 + 16 + hs * 8];
    }
#pragma unroll
    for (int i = 0; i < 2; ++i)
#pragma unroll
      for (int j = 0; j < 4; ++j)
        acc[i][j] = __builtin_amdgcn_wmma_f32_16x16x32_bf16(
            false, af[i].v, false, bfr[j].v, (short)0, acc[i][j], false, false);
  }

  // epilogue: D 16x16 f32 layout -> lane holds col N=(lane&15), rows M=vgpr(+8 if lane>=16)
  const int r0  = (lane >> 4) * 8;
  const int col = lane & 15;
#pragma unroll
  for (int j = 0; j < 4; ++j) {
    const int dcol = n0 + wn * 64 + j * 16 + col;
    float bv = 0.f;
    if constexpr (EPI == 1 || EPI == 3) bv = bias[dcol];
    float s = 0.f, s2 = 0.f;
#pragma unroll
    for (int i = 0; i < 2; ++i) {
      const int mrow = m0 + wm * 32 + i * 16 + r0;
#pragma unroll
      for (int rr = 0; rr < 8; ++rr) {
        float v = acc[i][j][rr] + bv;
        if constexpr (EPI == 0 || EPI == 1)
          ((__bf16*)D)[batchD + (size_t)(mrow + rr) * ND + dcol] = (__bf16)v;
        else
          ((float*)D)[batchD + (size_t)(mrow + rr) * ND + dcol] = v;
        s += v;
        s2 += v * v;
      }
    }
    if constexpr (EPI == 3) {
      s  += __shfl_xor(s, 16, 32);
      s2 += __shfl_xor(s2, 16, 32);
      if (lane < 16) {
        atomicAdd(&csum[dcol], s);
        atomicAdd(&csumsq[dcol], s2);
      }
    }
  }
}

// ---------------------------------------------------------------------------
// Softmax over tokens (dim N) per (b, k) column: online max + sumexp.
// logits [B, N_, K_]; block handles (b, 32 k-columns); tx = k-lane (coalesced).
// ---------------------------------------------------------------------------
__global__ __launch_bounds__(256) void softmax_col_stats(const float* __restrict__ logits,
                                                         float* __restrict__ cmax,
                                                         float* __restrict__ csum) {
  __shared__ float sm[8][32], ss[8][32];
  const int tx = threadIdx.x & 31;
  const int ty = threadIdx.x >> 5;
  const int b  = blockIdx.x >> 2;      // K_/32 == 4 groups
  const int kg = blockIdx.x & 3;
  const int k  = kg * 32 + tx;
  const float* base = logits + (size_t)b * N_ * K_ + k;
  float m = -__builtin_inff(), s = 0.f;
  for (int n = ty; n < N_; n += 8) {
    float v  = base[(size_t)n * K_];
    float nm = fmaxf(m, v);
    s = s * __expf(m - nm) + __expf(v - nm);
    m = nm;
  }
  sm[ty][tx] = m;
  ss[ty][tx] = s;
  __syncthreads();
  if (ty == 0) {
    float M = m, S = s;
#pragma unroll
    for (int i = 1; i < 8; ++i) {
      float mi = sm[i][tx], si = ss[i][tx];
      float nm = fmaxf(M, mi);
      S = S * __expf(M - nm) + si * __expf(mi - nm);
      M = nm;
    }
    cmax[b * K_ + k] = M;
    csum[b * K_ + k] = S;
  }
}

// ---------------------------------------------------------------------------
// p = exp(a - colmax)/colsum ; p /= (1e-5 + sum_k p) ; store bf16.
// One wave (32 lanes) per token row: lane holds 4 consecutive k.
// ---------------------------------------------------------------------------
__global__ __launch_bounds__(256) void softmax_normalize(const float* __restrict__ logits,
                                                         const float* __restrict__ cmax,
                                                         const float* __restrict__ csum,
                                                         __bf16* __restrict__ attn) {
  const int lane = threadIdx.x & 31;
  const int row  = blockIdx.x * 8 + (threadIdx.x >> 5);
  const int b    = row >> 12;   // N_ == 4096
  const float4 a  = *(const float4*)(logits + (size_t)row * K_ + lane * 4);
  const float4 mM = *(const float4*)(cmax + b * K_ + lane * 4);
  const float4 sS = *(const float4*)(csum + b * K_ + lane * 4);
  float p0 = __expf(a.x - mM.x) / sS.x;
  float p1 = __expf(a.y - mM.y) / sS.y;
  float p2 = __expf(a.z - mM.z) / sS.z;
  float p3 = __expf(a.w - mM.w) / sS.w;
  float rs = p0 + p1 + p2 + p3;
#pragma unroll
  for (int off = 16; off > 0; off >>= 1) rs += __shfl_xor(rs, off, 32);
  const float inv = 1.0f / (1e-5f + rs);
  union { __bf16 h[4]; uint2 u; } pk;
  pk.h[0] = (__bf16)(p0 * inv);
  pk.h[1] = (__bf16)(p1 * inv);
  pk.h[2] = (__bf16)(p2 * inv);
  pk.h[3] = (__bf16)(p3 * inv);
  *(uint2*)(attn + (size_t)row * K_ + lane * 4) = pk.u;
}

// ---------------------------------------------------------------------------
// BN stats finalize: mean / rsqrt(var + 1e-3) per channel.
// ---------------------------------------------------------------------------
__global__ void bn_finalize(const float* __restrict__ csum, const float* __restrict__ csumsq,
                            float* __restrict__ mean, float* __restrict__ rstd) {
  const int d = threadIdx.x;
  const float cnt = (float)(B_ * N_);
  const float mu  = csum[d] / cnt;
  const float var = csumsq[d] / cnt - mu * mu;
  mean[d] = mu;
  rstd[d] = rsqrtf(var + 1e-3f);
}

// ---------------------------------------------------------------------------
// o3 [B, N_, C_] -> out [B, C_, N_]: BN affine + residual + LeakyReLU(0.2).
// Tiled LDS transpose so both global accesses stay coalesced.
// ---------------------------------------------------------------------------
__global__ __launch_bounds__(256) void bn_res_leaky_out(const float* __restrict__ o3,
                                                        const float* __restrict__ x,
                                                        const float* __restrict__ mean,
                                                        const float* __restrict__ rstd,
                                                        const float* __restrict__ gamma,
                                                        const float* __restrict__ beta,
                                                        float* __restrict__ out) {
  __shared__ float tile[32][33];
  const int tx = threadIdx.x, ty = threadIdx.y;
  const int n0 = blockIdx.x * 32, c0 = blockIdx.y * 32, b = blockIdx.z;
  const float* src = o3 + (size_t)b * N_ * C_;
#pragma unroll
  for (int j = 0; j < 4; ++j)
    tile[ty + 8 * j][tx] = src[(size_t)(n0 + ty + 8 * j) * C_ + c0 + tx];
  __syncthreads();
#pragma unroll
  for (int j = 0; j < 4; ++j) {
    const int c = c0 + ty + 8 * j;
    const int n = n0 + tx;
    const size_t xi = ((size_t)b * C_ + c) * N_ + n;
    float v = tile[tx][ty + 8 * j];
    v = (v - mean[c]) * rstd[c] * gamma[c] + beta[c] + x[xi];
    out[xi] = v >= 0.f ? v : 0.2f * v;
  }
}

// ---------------------------------------------------------------------------
extern "C" void kernel_launch(void* const* d_in, const int* in_sizes, int n_in,
                              void* d_out, int out_size, void* d_ws, size_t ws_size,
                              hipStream_t stream) {
  const float* x     = (const float*)d_in[0];
  const float* w1    = (const float*)d_in[1];
  const float* b1    = (const float*)d_in[2];
  const float* mk    = (const float*)d_in[3];
  const float* w2    = (const float*)d_in[4];
  const float* b2    = (const float*)d_in[5];
  const float* gamma = (const float*)d_in[6];
  const float* beta  = (const float*)d_in[7];
  float* out = (float*)d_out;

  char* ws = (char*)d_ws;
  size_t off = 0;
  auto alloc = [&](size_t bytes) {
    size_t o = off;
    off = (off + bytes + 255) & ~(size_t)255;
    return o;
  };
  const size_t oXT   = alloc((size_t)B_ * N_ * C_ * 2);  // xT bf16 (reused as o2 bf16)
  const size_t oO1   = alloc((size_t)B_ * N_ * C_ * 2);  // conv1 out bf16 [b,n,c]
  const size_t oLG   = alloc((size_t)B_ * N_ * K_ * 4);  // attn logits f32 [b,n,k]
  const size_t oAT   = alloc((size_t)B_ * N_ * K_ * 2);  // attn bf16 [b,n,k]
  const size_t oO3   = alloc((size_t)B_ * N_ * C_ * 4);  // conv2 out f32 [b,n,c]
  const size_t oW1   = alloc((size_t)C_ * C_ * 2);
  const size_t oMKT  = alloc((size_t)K_ * C_ * 2);       // mk^T bf16 [k,c]
  const size_t oMK   = alloc((size_t)C_ * K_ * 2);       // mk bf16 [c,k]
  const size_t oW2   = alloc((size_t)C_ * C_ * 2);
  const size_t oCMAX = alloc((size_t)B_ * K_ * 4);
  const size_t oCSUM = alloc((size_t)B_ * K_ * 4);
  const size_t oCHS  = alloc((size_t)C_ * 4);
  const size_t oCHS2 = alloc((size_t)C_ * 4);
  const size_t oMEAN = alloc((size_t)C_ * 4);
  const size_t oRSTD = alloc((size_t)C_ * 4);

  __bf16* xT     = (__bf16*)(ws + oXT);
  __bf16* o1b    = (__bf16*)(ws + oO1);
  float*  logits = (float*)(ws + oLG);
  __bf16* attnb  = (__bf16*)(ws + oAT);
  __bf16* o2b    = (__bf16*)(ws + oXT);   // alias: xT dead after conv1
  float*  o3     = (float*)(ws + oO3);
  __bf16* w1b    = (__bf16*)(ws + oW1);
  __bf16* mkTb   = (__bf16*)(ws + oMKT);
  __bf16* mkb    = (__bf16*)(ws + oMK);
  __bf16* w2b    = (__bf16*)(ws + oW2);
  float*  cmax   = (float*)(ws + oCMAX);
  float*  csumm  = (float*)(ws + oCSUM);
  float*  chs    = (float*)(ws + oCHS);
  float*  chs2   = (float*)(ws + oCHS2);
  float*  meanp  = (float*)(ws + oMEAN);
  float*  rstdp  = (float*)(ws + oRSTD);

  // --- stage 0: layout conversions (bf16) ---
  tconv_bf16<<<dim3(N_ / 32, C_ / 32, B_), dim3(32, 8), 0, stream>>>(
      x, xT, C_, N_, (long)C_ * N_, (long)N_ * C_);                 // xT[b,n,c]
  tconv_bf16<<<dim3(K_ / 32, C_ / 32, 1), dim3(32, 8), 0, stream>>>(
      mk, mkTb, C_, K_, 0, 0);                                      // mkT[k,c]
  f32_to_bf16<<<(C_ * C_ + 255) / 256, 256, 0, stream>>>(w1, w1b, C_ * C_);
  f32_to_bf16<<<(C_ * K_ + 255) / 256, 256, 0, stream>>>(mk, mkb, C_ * K_);
  f32_to_bf16<<<(C_ * C_ + 255) / 256, 256, 0, stream>>>(w2, w2b, C_ * C_);
  zero_f32<<<2, 256, 0, stream>>>(chs, C_);
  zero_f32<<<2, 256, 0, stream>>>(chs2, C_);

  // --- conv1: o1[b,n,d] = xT[b,n,:]*w1[d,:] + b1[d]  (bf16 out) ---
  gemm_bf16_wmma<C_, 1><<<dim3(N_ / 128, C_ / 128, B_), 256, 0, stream>>>(
      xT, w1b, b1, (void*)o1b, nullptr, nullptr, C_);

  // --- attn logits: lg[b,n,k] = o1[b,n,:]*mkT[k,:]  (f32 out) ---
  gemm_bf16_wmma<C_, 2><<<dim3(N_ / 128, K_ / 128, B_), 256, 0, stream>>>(
      o1b, mkTb, nullptr, (void*)logits, nullptr, nullptr, K_);

  // --- softmax over tokens + per-row renorm ---
  softmax_col_stats<<<B_ * (K_ / 32), 256, 0, stream>>>(logits, cmax, csumm);
  softmax_normalize<<<(B_ * N_) / 8, 256, 0, stream>>>(logits, cmax, csumm, attnb);

  // --- o2[b,n,c] = attn[b,n,:]*mk[c,:]  (bf16 out, aliases xT) ---
  gemm_bf16_wmma<K_, 0><<<dim3(N_ / 128, C_ / 128, B_), 256, 0, stream>>>(
      attnb, mkb, nullptr, (void*)o2b, nullptr, nullptr, C_);

  // --- conv2: o3[b,n,d] = o2[b,n,:]*w2[d,:] + b2[d]  (f32 out + BN stats) ---
  gemm_bf16_wmma<C_, 3><<<dim3(N_ / 128, C_ / 128, B_), 256, 0, stream>>>(
      o2b, w2b, b2, (void*)o3, chs, chs2, C_);

  // --- BN finalize + transpose + residual + LeakyReLU ---
  bn_finalize<<<1, C_, 0, stream>>>(chs, chs2, meanp, rstdp);
  bn_res_leaky_out<<<dim3(N_ / 32, C_ / 32, B_), dim3(32, 8), 0, stream>>>(
      o3, x, meanp, rstdp, gamma, beta, out);
}
